// model_12773232738425
// MI455X (gfx1250) — matrix operand order repaired and checked by reading, not compiled, not run on hardware
//
#include <hip/hip_runtime.h>
#include <cstdint>
#include <cstddef>

// ---------------------------------------------------------------------------
// Types for CDNA5 WMMA (wave32): bf16 A/B fragments, f32 accumulator.
// ---------------------------------------------------------------------------
typedef __attribute__((ext_vector_type(16))) __bf16       v16bf;
typedef __attribute__((ext_vector_type(8)))  float        v8f;
typedef __attribute__((ext_vector_type(8)))  unsigned int v8u;
typedef __attribute__((ext_vector_type(4)))  unsigned int v4u;

#define BN_EPS 1e-5f

__device__ __forceinline__ int iminD(int a, int b) { return a < b ? a : b; }
__device__ __forceinline__ int imaxD(int a, int b) { return a > b ? a : b; }
__device__ __forceinline__ float bf2f(unsigned short s) {
  return (float)__builtin_bit_cast(__bf16, s);
}
__device__ __forceinline__ unsigned short f2bfbits(float f) {
  return __builtin_bit_cast(unsigned short, (__bf16)f);
}

// ---------------------------------------------------------------------------
// Implicit-GEMM conv3d with fused BN + residual + activation.
//   GEMM: M = cout, N = out voxels, K = ntaps * cpad (tap-major, cpad=rup16(cin))
//   One wave -> 16(M) x 64(N): 4 accumulators share each A fragment.
//
// Data layout: activations are NDHWC bf16, [voxel][channels], channels padded
// to a multiple of 16 (zero-filled where the logical channel count is smaller).
// Hot loop per 32-K block:
//   * A fragment: two unconditional global_load_b128 from padded weights.
//   * ALL FOUR B fragments loaded into distinct registers first (16 consecutive
//     channels at one voxel -> two global_load_b128 from `sval ? bp : zp`,
//     32-bit offset math), then 4 back-to-back WMMAs: loads overlap, no WAR
//     hazard padding between fragments.
//   * One tap decompose + one validity bit per fragment (27-bit lane mask).
//   * K-tail reads are neutralized by zero-padded weights.
//
// VGPR layouts per CDNA5 ISA 7.12.2:
//   A 16x32 bf16 : lane l (m=l%16), elem e=2j+p holds
//                  k = e + 16*(l>=16)  (as loaded here, same order as B)
//   B 32x16 bf16 : lane l (n=l%16), elem e holds k = e + 16*(l>=16) (this code's order, not the ISA table)
//   C/D 16x16 f32: lane l (n=l%16), vgpr j holds m = j + 8*(l>=16)
// ---------------------------------------------------------------------------
struct ConvArgs {
  const __bf16* __restrict__ x;     // [in_d^3][cpad] bf16 (NDHWC, padded)
  const __bf16* __restrict__ w;     // [coutPad][Kpad] bf16 (tap-major, padded)
  const __bf16* __restrict__ zp;    // 32B zero page
  const float*  g;                  // BN gamma (null -> no BN scale)
  const float*  beta;               // BN beta / bias (null -> none)
  const __bf16* res;                // residual [out_d^3][coutC] (null -> none)
  __bf16*       y;                  // bf16 out [out_d^3][coutC] (optional)
  float*        yf;                 // f32 out, planar [cout][out_d^3] (optional)
  int cin, cout, coutC, cpad, Kpad; // Kpad = rup32(ntaps*cpad) = w row stride
  int ntaps;                        // 27 (3x3x3) or 1 (1x1x1)
  int cp16;                         // cpad >> 4
  float invCp16;                    // 1.0f / cp16
  int in_d, out_d, stride, pad, act;  // act: 0 none, 1 relu, 2 leaky
};

__global__ __launch_bounds__(256) void conv3d_wmma_kernel(ConvArgs a) {
  const int  lane = threadIdx.x & 31;
  const long wave = (long)blockIdx.x * 8 + (threadIdx.x >> 5);
  const int  od   = a.out_d;
  const int  nvox = od * od * od;
  const int  nTN  = (nvox + 63) >> 6;       // 4 N-subtiles of 16 per wave
  const int  nTM  = (a.cout + 15) >> 4;
  if (wave >= (long)nTM * (long)nTN) return;  // wave-uniform exit

  const int mT   = (int)(wave / nTN);
  const int nT   = (int)(wave - (long)mT * nTN);
  const int half = lane >> 4;
  const int lm   = lane & 15;
  const int mBase = mT << 4;
  const int nBase = nT << 6;

  const int id = a.in_d, inPlane = id * id;

  // Per-lane, per-subtile spatial setup + 27-bit tap validity masks.
  int iz0[4], iy0[4], ix0[4], nIdx[4];
  bool nOK[4];
  unsigned tmask[4];
#pragma unroll
  for (int t = 0; t < 4; ++t) {
    const int nr = nBase + 16 * t + lm;
    nOK[t] = nr < nvox;
    const int n = nOK[t] ? nr : (nvox - 1);   // clamp: loads stay in-bounds
    nIdx[t] = n;
    const int oz = n / (od * od);
    const int r  = n - oz * od * od;
    const int oy = r / od;
    const int ox = r - oy * od;
    iz0[t] = oz * a.stride - a.pad;
    iy0[t] = oy * a.stride - a.pad;
    ix0[t] = ox * a.stride - a.pad;
    unsigned m = 0;
    if (a.ntaps == 1) {
      m = 1u;                                  // 1x1x1 pad0: always in range
    } else {
      int tap = 0;
#pragma unroll
      for (int dz = 0; dz < 3; ++dz)
#pragma unroll
        for (int dy = 0; dy < 3; ++dy)
#pragma unroll
          for (int dxk = 0; dxk < 3; ++dxk, ++tap) {
            const bool ok = (unsigned)(iz0[t] + dz) < (unsigned)id &&
                            (unsigned)(iy0[t] + dy) < (unsigned)id &&
                            (unsigned)(ix0[t] + dxk) < (unsigned)id;
            m |= (ok ? 1u : 0u) << tap;
          }
    }
    tmask[t] = m;
  }

  const __bf16* wrow = a.w + (size_t)(mBase + lm) * a.Kpad;  // row exists: padded

  v8f acc0 = {}, acc1 = {}, acc2 = {}, acc3 = {};

  for (int kb = 0; kb < a.Kpad; kb += 32) {
    const int k0  = kb + 16 * half;
    const int t16 = k0 >> 4;
    int tap = (int)(((float)t16 + 0.5f) * a.invCp16);   // exact for our ranges
    const int ci0 = (t16 - tap * a.cp16) << 4;
    if (tap >= a.ntaps) tap = a.ntaps - 1;              // zero weights cover tail
    int dz = 0, dy = 0, dxk = 0;
    if (a.ntaps != 1) {
      dz = tap / 9; const int rr = tap - dz * 9;
      dy = rr / 3;  dxk = rr - dy * 3;
    }

    // ---- A fragment: two unconditional 16B loads ----
    v8u au;
    {
      const v4u q0 = *(const v4u*)(wrow + kb + 16 * half);
      const v4u q1 = *(const v4u*)(wrow + kb + 16 * half + 8);
      au[0] = q0[0]; au[1] = q0[1]; au[2] = q0[2]; au[3] = q0[3];
      au[4] = q1[0]; au[5] = q1[1]; au[6] = q1[2]; au[7] = q1[3];
    }
    const v16bf av = __builtin_bit_cast(v16bf, au);

    // ---- load ALL 4 B fragments first (distinct regs, overlapped loads) ----
    v16bf bvv[4];
#pragma unroll
    for (int t = 0; t < 4; ++t) {
      const int iz = iminD(imaxD(iz0[t] + dz, 0), id - 1);
      const int iy = iminD(imaxD(iy0[t] + dy, 0), id - 1);
      const int ix = iminD(imaxD(ix0[t] + dxk, 0), id - 1);
      const unsigned sval = (tmask[t] >> tap) & 1u;
      const unsigned off32 =
          (unsigned)((iz * inPlane + iy * id + ix) * a.cpad + ci0);  // fits u32
      const __bf16* bp  = a.x + off32;
      const __bf16* sel = sval ? bp : a.zp;     // v_cndmask on address
      const v4u b0 = *(const v4u*)sel;
      const v4u b1 = *(const v4u*)(sel + 8);
      v8u bu;
      bu[0] = b0[0]; bu[1] = b0[1]; bu[2] = b0[2]; bu[3] = b0[3];
      bu[4] = b1[0]; bu[5] = b1[1]; bu[6] = b1[2]; bu[7] = b1[3];
      bvv[t] = __builtin_bit_cast(v16bf, bu);
    }

    // ---- 4 back-to-back WMMAs (no shared regs besides A) ----
    acc0 = __builtin_amdgcn_wmma_f32_16x16x32_bf16(false, av, false, bvv[0],
                                                   (short)0, acc0, false, false);
    acc1 = __builtin_amdgcn_wmma_f32_16x16x32_bf16(false, av, false, bvv[1],
                                                   (short)0, acc1, false, false);
    acc2 = __builtin_amdgcn_wmma_f32_16x16x32_bf16(false, av, false, bvv[2],
                                                   (short)0, acc2, false, false);
    acc3 = __builtin_amdgcn_wmma_f32_16x16x32_bf16(false, av, false, bvv[3],
                                                   (short)0, acc3, false, false);
  }

  // ---- fused epilogue: BN scale/bias, residual, activation ----
  const int coutC = a.coutC;
  const float sBN = rsqrtf(1.f + BN_EPS);
  float scale[8], bias0[8];
#pragma unroll
  for (int j = 0; j < 8; ++j) {
    const int m  = mBase + j + half * 8;
    const int mc = iminD(m, a.cout - 1);
    scale[j] = 1.f; bias0[j] = 0.f;
    if (a.g)         { scale[j] = a.g[mc] * sBN; bias0[j] = a.beta[mc]; }
    else if (a.beta) { bias0[j] = a.beta[mc]; }
  }
#pragma unroll
  for (int t = 0; t < 4; ++t) {
    const unsigned oOff = (unsigned)(nIdx[t] * coutC) + mBase + 8 * half;
    float rv[8] = {0.f, 0.f, 0.f, 0.f, 0.f, 0.f, 0.f, 0.f};
    if (a.res) {
      const v4u r = *(const v4u*)(a.res + oOff);
#pragma unroll
      for (int dwi = 0; dwi < 4; ++dwi) {
        rv[2 * dwi]     = bf2f((unsigned short)(r[dwi] & 0xffffu));
        rv[2 * dwi + 1] = bf2f((unsigned short)(r[dwi] >> 16));
      }
    }
    float vals[8];
#pragma unroll
    for (int j = 0; j < 8; ++j) {
      float val = (t == 0) ? acc0[j] : (t == 1) ? acc1[j] : (t == 2) ? acc2[j]
                                                                     : acc3[j];
      val = val * scale[j] + bias0[j] + rv[j];
      if (a.act == 1)      val = fmaxf(val, 0.f);
      else if (a.act == 2) val = (val > 0.f) ? val : 0.2f * val;
      vals[j] = val;
    }
    if (a.y && nOK[t]) {
      v4u o;
#pragma unroll
      for (int dwi = 0; dwi < 4; ++dwi)
        o[dwi] = (unsigned)f2bfbits(vals[2 * dwi]) |
                 ((unsigned)f2bfbits(vals[2 * dwi + 1]) << 16);
      *(v4u*)(a.y + oOff) = o;  // 16B store
    }
    if (a.yf) {
#pragma unroll
      for (int j = 0; j < 8; ++j) {
        const int m = mBase + j + half * 8;
        if (m < a.cout && nOK[t])
          a.yf[(unsigned)(m * nvox) + (unsigned)nIdx[t]] = vals[j];
      }
    }
  }
}

// ---------------------------------------------------------------------------
// Weight f32 OIDHW -> bf16 [coutPad][Kpad], tap-major K (k = tap*cpad + ci),
// zeros for co>=cout, tap>=ntaps, ci>=cin.
// ---------------------------------------------------------------------------
__global__ void wconv_kernel(const float* __restrict__ w, __bf16* __restrict__ out,
                             int cout, int cin, int ntaps, int cpad,
                             int coutPad, int Kpad) {
  const long i = (long)blockIdx.x * 256 + threadIdx.x;
  const long total = (long)coutPad * Kpad;
  if (i >= total) return;
  const int co  = (int)(i / Kpad);
  const int k   = (int)(i - (long)co * Kpad);
  const int tap = k / cpad;
  const int ci  = k - tap * cpad;
  float v = 0.f;
  if (co < cout && tap < ntaps && ci < cin)
    v = w[((size_t)co * cin + ci) * ntaps + tap];
  out[i] = (__bf16)v;
}

// Zero-fill a bf16 region (zero page init).
__global__ void zero_fill_kernel(__bf16* __restrict__ p, long count) {
  const long i = (long)blockIdx.x * 256 + threadIdx.x;
  if (i < count) p[i] = (__bf16)0.f;
}

// source(1ch) + target_volume(4ch) -> NDHWC bf16 [nv][16] (channels 5..15 zero)
__global__ void pack_input_kernel(const float* __restrict__ src,
                                  const float* __restrict__ tv,
                                  __bf16* __restrict__ out, int nv) {
  const long i = (long)blockIdx.x * 256 + threadIdx.x;
  if (i >= (long)nv * 16) return;
  const int v = (int)(i >> 4);
  const int c = (int)(i & 15);
  float x = 0.f;
  if (c == 0)      x = src[v];
  else if (c <= 4) x = tv[(size_t)(c - 1) * nv + v];
  out[i] = (__bf16)x;
}

// 2x trilinear upsample (half-pixel centers), NDHWC bf16 -> NDHWC (strided out).
// C = 1<<cShift (in channels), catC = out channel stride (>= C).
__global__ void upsample_kernel(const __bf16* __restrict__ in, __bf16* __restrict__ out,
                                int cShift, int catC, int idim) {
  const int  od  = idim * 2;
  const long onv = (long)od * od * od;
  const long i   = (long)blockIdx.x * 256 + threadIdx.x;
  if (i >= (onv << cShift)) return;
  const long ov = i >> cShift;
  const int  c  = (int)(i & ((1 << cShift) - 1));
  const int o0 = (int)(ov / (od * od));
  const int rr = (int)(ov - (long)o0 * od * od);
  const int o1 = rr / od;
  const int o2 = rr - o1 * od;
  const float s0 = (o0 + 0.5f) * 0.5f - 0.5f;
  const float s1 = (o1 + 0.5f) * 0.5f - 0.5f;
  const float s2 = (o2 + 0.5f) * 0.5f - 0.5f;
  int a0 = (int)floorf(s0), a1 = (int)floorf(s1), a2 = (int)floorf(s2);
  const float f0 = s0 - a0, f1 = s1 - a1, f2 = s2 - a2;
  int b0 = a0 + 1, b1 = a1 + 1, b2 = a2 + 1;
  a0 = iminD(imaxD(a0, 0), idim - 1); b0 = iminD(imaxD(b0, 0), idim - 1);
  a1 = iminD(imaxD(a1, 0), idim - 1); b1 = iminD(imaxD(b1, 0), idim - 1);
  a2 = iminD(imaxD(a2, 0), idim - 1); b2 = iminD(imaxD(b2, 0), idim - 1);
  const int ip = idim * idim;
  const int C  = 1 << cShift;
  const __bf16* p = in + c;
  const float v000 = (float)p[(size_t)(a0 * ip + a1 * idim + a2) * C];
  const float v001 = (float)p[(size_t)(a0 * ip + a1 * idim + b2) * C];
  const float v010 = (float)p[(size_t)(a0 * ip + b1 * idim + a2) * C];
  const float v011 = (float)p[(size_t)(a0 * ip + b1 * idim + b2) * C];
  const float v100 = (float)p[(size_t)(b0 * ip + a1 * idim + a2) * C];
  const float v101 = (float)p[(size_t)(b0 * ip + a1 * idim + b2) * C];
  const float v110 = (float)p[(size_t)(b0 * ip + b1 * idim + a2) * C];
  const float v111 = (float)p[(size_t)(b0 * ip + b1 * idim + b2) * C];
  const float c00 = v000 * (1.f - f2) + v001 * f2;
  const float c01 = v010 * (1.f - f2) + v011 * f2;
  const float c10 = v100 * (1.f - f2) + v101 * f2;
  const float c11 = v110 * (1.f - f2) + v111 * f2;
  const float c0 = c00 * (1.f - f1) + c01 * f1;
  const float c1 = c10 * (1.f - f1) + c11 * f1;
  out[ov * catC + c] = (__bf16)(c0 * (1.f - f0) + c1 * f0);
}

// Interleave skip channels into concat buffer: out[v][c1+c] = skip[v][c].
// C = 1<<cShift.
__global__ void concat_copy_kernel(const __bf16* __restrict__ skip,
                                   __bf16* __restrict__ out,
                                   int cShift, int catC, int c1, long nv) {
  const long i = (long)blockIdx.x * 256 + threadIdx.x;
  if (i >= (nv << cShift)) return;
  const long v = i >> cShift;
  const int  c = (int)(i & ((1 << cShift) - 1));
  out[v * catC + c1 + c] = skip[i];
}

// phi = disp * 2/(S-1) + identity_map   (f32 planar)
__global__ void phi_kernel(const float* __restrict__ disp, float* __restrict__ phi, int n) {
  const long nv = (long)n * n * n;
  const long i  = (long)blockIdx.x * 256 + threadIdx.x;
  if (i >= 3 * nv) return;
  const int c = (int)(i / nv);
  const long v = i - (long)c * nv;
  const int i0 = (int)(v / (n * n));
  const int r  = (int)(v - (long)i0 * n * n);
  const int i1 = r / n;
  const int i2 = r - i1 * n;
  const int a = (c == 0) ? i0 : ((c == 1) ? i1 : i2);
  const float idv = -1.f + 2.f * (float)a / (float)(n - 1);
  phi[i] = disp[i] * (2.f / (float)(n - 1)) + idv;
}

// trilinear warp of density by phi (reference semantics incl. (v+1)/2 mapping)
__global__ void warp_kernel(const float* __restrict__ density, const float* __restrict__ phi,
                            float* __restrict__ out, int n) {
  const long nv = (long)n * n * n;
  const long v  = (long)blockIdx.x * 256 + threadIdx.x;
  if (v >= nv) return;
  const float px = phi[v], py = phi[nv + v], pz = phi[2 * nv + v];
  const float cx = (px + 1.f) * 0.5f * (n - 1);
  const float cy = (py + 1.f) * 0.5f * (n - 1);
  const float cz = (pz + 1.f) * 0.5f * (n - 1);
  const float x0f = floorf(cx), y0f = floorf(cy), z0f = floorf(cz);
  const float fx = cx - x0f, fy = cy - y0f, fz = cz - z0f;
  const int x0 = (int)x0f, y0 = (int)y0f, z0 = (int)z0f;
  float acc = 0.f;
#pragma unroll
  for (int dx = 0; dx < 2; ++dx)
#pragma unroll
    for (int dy = 0; dy < 2; ++dy)
#pragma unroll
      for (int dz = 0; dz < 2; ++dz) {
        const float w = (dx ? fx : 1.f - fx) * (dy ? fy : 1.f - fy) * (dz ? fz : 1.f - fz);
        const int ix = x0 + dx, iy = y0 + dy, iz = z0 + dz;
        float val = 0.f;
        if ((unsigned)ix < (unsigned)n && (unsigned)iy < (unsigned)n &&
            (unsigned)iz < (unsigned)n)
          val = (density[(size_t)ix * n * n + iy * n + iz] + 1.f) * 0.5f;
        acc += w * val;
      }
  out[v] = acc * 2.f - 1.f;
}

// Siddon ray cast: incremental merge of per-axis plane crossings
// (equivalent to reference's clip+sort of all plane alphas).
__global__ void siddon_kernel(const float* __restrict__ vol, const float* __restrict__ pose,
                              const float* __restrict__ affine, float* __restrict__ proj) {
  const int rid = blockIdx.x * blockDim.x + threadIdx.x;
  if (rid >= 384 * 384) return;
  const int ih = rid / 384, iw = rid - 384 * ih;
  const float y = -((float)iw - 191.5f) * 0.7255f;  // DELX
  const float z =  ((float)ih - 191.5f) * 0.7255f;  // DELY
  // detector coords; REORIENT swaps y<->z
  float t1[3] = { 510.0f, z, y };
  float s1[3] = { -510.0f, 0.f, 0.f };
  // pose (row-major 4x4): p' = R p + t
  float tw[3], sw[3];
#pragma unroll
  for (int i = 0; i < 3; ++i) {
    tw[i] = pose[i*4+0]*t1[0] + pose[i*4+1]*t1[1] + pose[i*4+2]*t1[2] + pose[i*4+3];
    sw[i] = pose[i*4+0]*s1[0] + pose[i*4+1]*s1[1] + pose[i*4+2]*s1[2] + pose[i*4+3];
  }
  const float rl0 = tw[0]-sw[0], rl1 = tw[1]-sw[1], rl2 = tw[2]-sw[2];
  const float raylen = sqrtf(rl0*rl0 + rl1*rl1 + rl2*rl2);
  // inverse of affine (4x4 with last row 0 0 0 1): inv3x3 via adjugate
  float M[9];
#pragma unroll
  for (int i = 0; i < 3; ++i)
#pragma unroll
    for (int j = 0; j < 3; ++j) M[i*3+j] = affine[i*4+j];
  const float tv0 = affine[3], tv1 = affine[7], tv2 = affine[11];
  const float det = M[0]*(M[4]*M[8]-M[5]*M[7]) - M[1]*(M[3]*M[8]-M[5]*M[6])
                  + M[2]*(M[3]*M[7]-M[4]*M[6]);
  const float idet = 1.0f / det;
  float inv[9];
  inv[0]=(M[4]*M[8]-M[5]*M[7])*idet; inv[1]=(M[2]*M[7]-M[1]*M[8])*idet; inv[2]=(M[1]*M[5]-M[2]*M[4])*idet;
  inv[3]=(M[5]*M[6]-M[3]*M[8])*idet; inv[4]=(M[0]*M[8]-M[2]*M[6])*idet; inv[5]=(M[2]*M[3]-M[0]*M[5])*idet;
  inv[6]=(M[3]*M[7]-M[4]*M[6])*idet; inv[7]=(M[1]*M[6]-M[0]*M[7])*idet; inv[8]=(M[0]*M[4]-M[1]*M[3])*idet;
  float src[3], tgt[3];
#pragma unroll
  for (int i = 0; i < 3; ++i) {
    const float ti = -(inv[i*3+0]*tv0 + inv[i*3+1]*tv1 + inv[i*3+2]*tv2);
    src[i] = inv[i*3+0]*sw[0] + inv[i*3+1]*sw[1] + inv[i*3+2]*sw[2] + ti;
    tgt[i] = inv[i*3+0]*tw[0] + inv[i*3+1]*tw[1] + inv[i*3+2]*tw[2] + ti;
  }
  const int N = 96;
  float d[3], ds[3];
#pragma unroll
  for (int i = 0; i < 3; ++i) {
    d[i]  = tgt[i] - src[i];
    ds[i] = (fabsf(d[i]) < 1e-8f) ? 1e-8f : d[i];
  }
  float amin = 0.f, amax = 1.f;
#pragma unroll
  for (int i = 0; i < 3; ++i) {
    const float a0 = (-0.5f - src[i]) / ds[i];
    const float a1 = ((float)N - 0.5f - src[i]) / ds[i];
    amin = fmaxf(amin, fminf(a0, a1));
    amax = fminf(amax, fmaxf(a0, a1));
  }
  float out = 0.f;
  if (amax > amin) {
    float nexta[3], dal[3];
#pragma unroll
    for (int i = 0; i < 3; ++i) {
      dal[i] = fabsf(1.0f / ds[i]);
      const float kf = src[i] + 0.5f + amin * ds[i];  // continuous plane index at amin
      float k0;
      if (ds[i] > 0.f) k0 = floorf(kf) + 1.0f;
      else             k0 = ceilf(kf) - 1.0f;
      float a = ((k0 - 0.5f) - src[i]) / ds[i];
      if (k0 < 0.f || k0 > (float)N || a >= amax) a = 1e30f;
      nexta[i] = a;
    }
    float acur = amin;
    for (int it = 0; it < 3 * (N + 1) + 4; ++it) {
      const float an = fminf(fminf(nexta[0], nexta[1]), fminf(nexta[2], amax));
      const float step = an - acur;
      if (step > 0.f) {
        const float mid = 0.5f * (acur + an);
        int vx = (int)floorf(src[0] + mid * d[0] + 0.5f);
        int vy = (int)floorf(src[1] + mid * d[1] + 0.5f);
        int vz = (int)floorf(src[2] + mid * d[2] + 0.5f);
        vx = iminD(imaxD(vx, 0), N - 1);
        vy = iminD(imaxD(vy, 0), N - 1);
        vz = iminD(imaxD(vz, 0), N - 1);
        out += vol[(size_t)vx * N * N + vy * N + vz] * step;
      }
      if (an >= amax) break;
#pragma unroll
      for (int i = 0; i < 3; ++i)
        if (nexta[i] == an) nexta[i] += dal[i];
      acur = an;
    }
  }
  proj[rid] = out * raylen;
}

// ---------------------------------------------------------------------------
// Host driver.
// Input index map (setup_inputs insertion order; params nested in insertion
// order: w1,g1,b1,w2,g2,b2[,ws,gs,bs]):
//   0 source 1 target_volume 2 density 3 target_proj 4 target_poses 5 affine
//   6..14 init, 15..23 enc0, 24..32 enc1, 33..41 enc2, 42..50 enc3, 51..59 enc4
//   60..65 bottleneck, 66..74 dec0, 75..83 dec1, 84..92 dec2, 93..101 dec3,
//   102..110 dec4, 111 out_w, 112 out_b
// d_out: [warped 884736][phi 2654208][disp 2654208][proj 147456]
// ---------------------------------------------------------------------------
extern "C" void kernel_launch(void* const* d_in, const int* in_sizes, int n_in,
                              void* d_out, int out_size, void* d_ws, size_t ws_size,
                              hipStream_t stream) {
  (void)in_sizes; (void)n_in; (void)out_size; (void)ws_size;
  const long n0 = 884736, n1 = 110592, n2 = 13824, n3 = 1728, n4 = 216, n5 = 27;

  struct CD { int inIdx, cout, cin, ksz; };
  static const CD convs[36] = {
    {6,16,5,3},{9,16,16,3},{12,16,5,1},
    {15,32,16,3},{18,32,32,3},{21,32,16,1},
    {24,64,32,3},{27,64,64,3},{30,64,32,1},
    {33,128,64,3},{36,128,128,3},{39,128,64,1},
    {42,128,128,3},{45,128,128,3},{48,128,128,1},
    {51,128,128,3},{54,128,128,3},{57,128,128,1},
    {60,128,128,3},{63,128,128,3},
    {66,128,256,3},{69,128,128,3},{72,128,256,1},
    {75,128,256,3},{78,128,128,3},{81,128,256,1},
    {84,64,192,3},{87,64,64,3},{90,64,192,1},
    {93,32,96,3},{96,32,32,3},{99,32,96,1},
    {102,16,48,3},{105,16,16,3},{108,16,48,1},
    {111,3,16,3},
  };

  char* wsp = (char*)d_ws;
  size_t off = 0;
  auto alloc = [&](size_t bytes) -> void* {
    off = (off + 255) & ~(size_t)255;
    void* p = wsp + off;
    off += bytes;
    return p;
  };

  __bf16* wb[36]; int cpadA[36], kpadA[36], coutPadA[36];
  for (int i = 0; i < 36; ++i) {
    const int ntaps = convs[i].ksz * convs[i].ksz * convs[i].ksz;
    const int cpad  = (convs[i].cin + 15) & ~15;
    const int Kpad  = (ntaps * cpad + 31) & ~31;
    const int cp    = (convs[i].cout + 15) & ~15;
    cpadA[i] = cpad; kpadA[i] = Kpad; coutPadA[i] = cp;
    wb[i] = (__bf16*)alloc((size_t)cp * Kpad * sizeof(__bf16));
  }
  // NDHWC activation buffers: [voxels][channels], channels already mult of 16
  // except the 5-channel input (padded to 16, zero-filled in pack).
  __bf16* zp    = (__bf16*)alloc(64);                       // 32B zero page
  __bf16* xin0  = (__bf16*)alloc((size_t)16 * n0 * 2);
  __bf16* skip0 = (__bf16*)alloc((size_t)16 * n0 * 2);
  __bf16* skip1 = (__bf16*)alloc((size_t)32 * n1 * 2);
  __bf16* skip2 = (__bf16*)alloc((size_t)64 * n2 * 2);
  __bf16* skip3 = (__bf16*)alloc((size_t)128 * n3 * 2);
  __bf16* skip4 = (__bf16*)alloc((size_t)128 * n4 * 2);
  __bf16* x5    = (__bf16*)alloc((size_t)128 * n5 * 2);
  __bf16* xbo   = (__bf16*)alloc((size_t)128 * n5 * 2);
  __bf16* catb  = (__bf16*)alloc((size_t)48 * n0 * 2);
  __bf16* t1b   = (__bf16*)alloc((size_t)16 * n0 * 2);
  __bf16* tres  = (__bf16*)alloc((size_t)16 * n0 * 2);
  __bf16* xdec  = (__bf16*)alloc((size_t)16 * n0 * 2);

  zero_fill_kernel<<<1, 256, 0, stream>>>(zp, 32);

  // Convert/permute all conv weights to padded, tap-major bf16.
  for (int i = 0; i < 36; ++i) {
    const int ntaps = convs[i].ksz * convs[i].ksz * convs[i].ksz;
    const long tot = (long)coutPadA[i] * kpadA[i];
    wconv_kernel<<<(int)((tot + 255) / 256), 256, 0, stream>>>(
        (const float*)d_in[convs[i].inIdx], wb[i], convs[i].cout, convs[i].cin,
        ntaps, cpadA[i], coutPadA[i], kpadA[i]);
  }

  auto F = [&](int i) -> const float* { return (const float*)d_in[i]; };

  auto conv = [&](int cid, const __bf16* x, const float* g, const float* be,
                  const __bf16* res, __bf16* y, float* yf, int in_d, int stride,
                  int act) {
    const CD& c = convs[cid];
    const int pad = (c.ksz == 3) ? 1 : 0;
    const int out_d = (in_d + 2 * pad - c.ksz) / stride + 1;
    ConvArgs a;
    a.x = x; a.w = wb[cid]; a.zp = zp; a.g = g; a.beta = be; a.res = res;
    a.y = y; a.yf = yf;
    a.cin = c.cin; a.cout = c.cout; a.coutC = c.cout;
    a.cpad = cpadA[cid]; a.Kpad = kpadA[cid];
    a.ntaps = c.ksz * c.ksz * c.ksz;
    a.cp16 = cpadA[cid] >> 4;
    a.invCp16 = 1.0f / (float)(cpadA[cid] >> 4);
    a.in_d = in_d; a.out_d = out_d; a.stride = stride; a.pad = pad; a.act = act;
    const long nv = (long)out_d * out_d * out_d;
    const long waves = (long)((c.cout + 15) / 16) * ((nv + 63) / 64);
    conv3d_wmma_kernel<<<(int)((waves + 7) / 8), 256, 0, stream>>>(a);
  };

  auto rb = [&](int w1cid, int base, bool hasS, const __bf16* x, __bf16* out,
                int in_d, int stride, bool leaky) {
    const int out_d = (in_d + 2 - 3) / stride + 1;
    const __bf16* res = x;
    if (hasS) {
      conv(w1cid + 2, x, F(base + 7), F(base + 8), nullptr, tres, nullptr, in_d, stride, 0);
      res = tres;
    }
    conv(w1cid,     x,   F(base + 1), F(base + 2), nullptr, t1b, nullptr, in_d, stride,
         leaky ? 2 : 1);
    conv(w1cid + 1, t1b, F(base + 4), F(base + 5), res,     out, nullptr, out_d, 1,
         leaky ? 2 : 1);
  };

  pack_input_kernel<<<(int)((16 * n0 + 255) / 256), 256, 0, stream>>>(
      (const float*)d_in[0], (const float*)d_in[1], xin0, (int)n0);

  // Encoder (outputs are the skip buffers)
  rb(0,  6,  true,  xin0,  skip0, 96, 1, false);
  rb(3,  15, true,  skip0, skip1, 96, 2, false);
  rb(6,  24, true,  skip1, skip2, 48, 2, false);
  rb(9,  33, true,  skip2, skip3, 24, 2, false);
  rb(12, 42, true,  skip3, skip4, 12, 2, false);
  rb(15, 51, true,  skip4, x5,    6,  2, false);
  rb(18, 60, false, x5,    xbo,   3,  1, false);

  // Decoder (curC and skip channels are powers of two -> shifts)
  const __bf16* skipsD[5] = { skip4, skip3, skip2, skip1, skip0 };
  const int skipShift[5] = {7, 7, 6, 5, 4};     // 128,128,64,32,16
  const int decCout[5]   = {128, 128, 64, 32, 16};
  const __bf16* cur = xbo; int curShift = 7; int curDim = 3;
  for (int i = 0; i < 5; ++i) {
    const int od = curDim * 2;
    const long onv = (long)od * od * od;
    const int curC = 1 << curShift;
    const int skC  = 1 << skipShift[i];
    const int catC = curC + skC;
    const long tot1 = onv << curShift;
    upsample_kernel<<<(int)((tot1 + 255) / 256), 256, 0, stream>>>(
        cur, catb, curShift, catC, curDim);
    const long tot2 = onv << skipShift[i];
    concat_copy_kernel<<<(int)((tot2 + 255) / 256), 256, 0, stream>>>(
        skipsD[i], catb, skipShift[i], catC, curC, onv);
    rb(20 + 3 * i, 66 + 9 * i, true, catb, xdec, od, 1, true);
    cur = xdec;
    curShift = (decCout[i] == 128) ? 7 : (decCout[i] == 64) ? 6
               : (decCout[i] == 32) ? 5 : 4;
    curDim = od;
  }

  float* dout = (float*)d_out;
  const long WARPED_OFF = 0;
  const long PHI_OFF  = 884736;
  const long DISP_OFF = 884736 + 2654208;
  const long PROJ_OFF = 884736 + 2 * 2654208;

  // Final 3-channel conv (bias only) -> disp (f32 planar, direct to d_out)
  conv(35, cur, nullptr, F(112), nullptr, nullptr, dout + DISP_OFF, 96, 1, 0);

  phi_kernel<<<(int)((3 * n0 + 255) / 256), 256, 0, stream>>>(
      dout + DISP_OFF, dout + PHI_OFF, 96);
  warp_kernel<<<(int)((n0 + 255) / 256), 256, 0, stream>>>(
      (const float*)d_in[2], dout + PHI_OFF, dout + WARPED_OFF, 96);
  siddon_kernel<<<(384 * 384 + 255) / 256, 256, 0, stream>>>(
      dout + WARPED_OFF, (const float*)d_in[4], (const float*)d_in[5],
      dout + PROJ_OFF);
}
